// PtrNet1_28252294873107
// MI455X (gfx1250) — compile-verified, matrix operand order checked by reading, not run on hardware
//
#include <hip/hip_runtime.h>
#include <hip/hip_bf16.h>
#include <cstdint>
#include <cstddef>

// ---------------- problem constants ----------------
#define B_    512
#define N_    512
#define FEAT_ 17
#define H_    128
#define G4_   512   // 4*HIDDEN
#define INF_  1e8f
#define CLIP_ 10.0f
#define GSTR  516   // LDS gate row stride (floats); 516 mod 64 = 4 -> row+8 lands 32 banks away

typedef __attribute__((ext_vector_type(16))) _Float16 v16h;
typedef __attribute__((ext_vector_type(8)))  float    v8f;

union FragU { v16h v; uint4 q[2]; _Float16 h[16]; };

// Load one 16x32 f16 WMMA fragment (A layout per ISA 7.12.2; B mirrored -> pass W's row n):
//   lane L: halves 0..7  = row[k0 + 8*(L>=16) + 0..7]
//           halves 8..15 = row[k0 + 8*(L>=16) + 16..23]
__device__ __forceinline__ v16h load_frag(const _Float16* __restrict__ rowk0, int lane) {
    FragU u;
    const _Float16* p = rowk0 + ((lane >> 4) << 3);
    u.q[0] = *reinterpret_cast<const uint4*>(p);
    u.q[1] = *reinterpret_cast<const uint4*>(p + 16);
    return u.v;
}

#define WMMA_F16(A, Bf, C) \
    __builtin_amdgcn_wmma_f32_16x16x32_f16(false, (A), false, (Bf), (short)0, (C), false, false)

// ---------------- packing / small kernels ----------------
__global__ void pack_wcat(const float* __restrict__ Wih, const float* __restrict__ Whh,
                          _Float16* __restrict__ Wcat) {
    int t = blockIdx.x * blockDim.x + threadIdx.x;          // 512*256
    if (t >= G4_ * 256) return;
    int n = t >> 8, k = t & 255;
    float v = (k < H_) ? Wih[n * H_ + k] : Whh[n * H_ + (k - H_)];
    Wcat[t] = (_Float16)v;
}

__global__ void add_bias_vec(const float* __restrict__ a, const float* __restrict__ b,
                             float* __restrict__ o, int n) {
    int t = blockIdx.x * blockDim.x + threadIdx.x;
    if (t < n) o[t] = a[t] + b[t];
}

__global__ void f32_to_f16_kernel(const float* __restrict__ a, _Float16* __restrict__ o, int n) {
    int t = blockIdx.x * blockDim.x + threadIdx.x;
    if (t < n) o[t] = (_Float16)a[t];
}

// emb[i, e] = sum_f x[i, f] * Wemb[e, f]   (K=17, pure VALU)
__global__ void embed_kernel(const float* __restrict__ x, const float* __restrict__ Wemb,
                             _Float16* __restrict__ emb, int rows) {
    long t = (long)blockIdx.x * blockDim.x + threadIdx.x;
    long i = t >> 7; int e = (int)(t & 127);
    if (i >= rows) return;
    const float* xr = x + i * FEAT_;
    const float* wr = Wemb + e * FEAT_;
    float s = 0.f;
    #pragma unroll
    for (int f = 0; f < FEAT_; ++f) s += xr[f] * wr[f];
    emb[t] = (_Float16)s;
}

__global__ void init_state(_Float16* __restrict__ h, float* __restrict__ c,
                           float* __restrict__ mask, float* __restrict__ ll,
                           int* __restrict__ pi) {
    int t = blockIdx.x * blockDim.x + threadIdx.x;          // 262144 threads
    if (t < B_ * H_) { h[t] = (_Float16)0.f; c[t] = 0.f; }
    if (t < B_ * N_) mask[t] = 0.f;
    if (t < B_)      ll[t] = 0.f;
    if (t < B_ * 3)  pi[t] = 0;
}

// ---------------- fused LSTM step ----------------
// gates(16x512 per block) = [x_t | h](16x256) * Wcat^T  via WMMA, staged in LDS,
// then the elementwise LSTM cell in the same kernel (block owns its 16 batch rows
// of h and c entirely, so no cross-block hazard).
// grid: 32 blocks x 256 thr (8 waves); wave = 16(M) x 64(N); K=256 -> 8 k-steps,
// software-pipelined (double-buffered fragments).
__global__ void lstm_step_fused(const _Float16* __restrict__ x, long xRowStride,
                                _Float16* __restrict__ h, float* __restrict__ c,
                                const _Float16* __restrict__ Wcat,   // 512 x 256 row-major
                                const float* __restrict__ bias,      // 512
                                _Float16* __restrict__ ref, int t) {
    __shared__ float gs[16 * GSTR];
    int tid   = threadIdx.x;
    int lane  = tid & 31;
    int mBase = blockIdx.x << 4;         // 16 batch rows per block
    int nBase = (tid >> 5) << 6;         // 64 gate cols per wave
    int arow  = mBase + (lane & 15);
    int ncol  = nBase + (lane & 15);

    const _Float16* xrow = x + (long)arow * xRowStride;
    const _Float16* hrow = h + (size_t)arow * H_;
    const _Float16* Wr0  = Wcat + (size_t)(ncol     ) * 256;
    const _Float16* Wr1  = Wcat + (size_t)(ncol + 16) * 256;
    const _Float16* Wr2  = Wcat + (size_t)(ncol + 32) * 256;
    const _Float16* Wr3  = Wcat + (size_t)(ncol + 48) * 256;

    v8f acc0 = {}, acc1 = {}, acc2 = {}, acc3 = {};
    v16h a  = load_frag(xrow, lane);
    v16h b0 = load_frag(Wr0, lane);
    v16h b1 = load_frag(Wr1, lane);
    v16h b2 = load_frag(Wr2, lane);
    v16h b3 = load_frag(Wr3, lane);
    #pragma unroll
    for (int kk = 0; kk < 8; ++kk) {
        v16h an, c0, c1, c2, c3;
        if (kk < 7) {                    // prefetch next k-slice before issuing WMMAs
            const int k1 = (kk + 1) << 5;
            if (k1 < 128) an = load_frag(xrow + k1, lane);
            else          an = load_frag(hrow + (k1 - 128), lane);
            c0 = load_frag(Wr0 + k1, lane);
            c1 = load_frag(Wr1 + k1, lane);
            c2 = load_frag(Wr2 + k1, lane);
            c3 = load_frag(Wr3 + k1, lane);
        }
        acc0 = WMMA_F16(a, b0, acc0);
        acc1 = WMMA_F16(a, b1, acc1);
        acc2 = WMMA_F16(a, b2, acc2);
        acc3 = WMMA_F16(a, b3, acc3);
        if (kk < 7) { a = an; b0 = c0; b1 = c1; b2 = c2; b3 = c3; }
    }

    // C layout (ISA 7.12.2): vgpr r, lane<16 -> row r, col lane; lane>=16 -> row r+8, col lane-16
    {
        int rrow = (lane >> 4) << 3;
        int ccol = lane & 15;
        #pragma unroll
        for (int r = 0; r < 8; ++r) {
            float* g = gs + (rrow + r) * GSTR + nBase + ccol;
            g[0]  = acc0[r];
            g[16] = acc1[r];
            g[32] = acc2[r];
            g[48] = acc3[r];
        }
    }
    __syncthreads();

    // elementwise LSTM cell: i,f,g,o gate order (torch layout)
    #pragma unroll
    for (int e = 0; e < 8; ++e) {
        int li  = tid + (e << 8);        // 0..2047 = 16 rows x 128 hid
        int row = li >> 7, hid = li & 127;
        int b   = mBase + row;
        const float* g = gs + row * GSTR;
        float gi = g[hid]          + bias[hid];
        float gf = g[H_   + hid]   + bias[H_   + hid];
        float gg = g[2*H_ + hid]   + bias[2*H_ + hid];
        float go = g[3*H_ + hid]   + bias[3*H_ + hid];
        float si = 1.f / (1.f + expf(-gi));
        float sf = 1.f / (1.f + expf(-gf));
        float so = 1.f / (1.f + expf(-go));
        long ci  = (long)b * H_ + hid;
        float cn = sf * c[ci] + si * tanhf(gg);
        float hn = so * tanhf(cn);
        c[ci] = cn;
        h[ci] = (_Float16)hn;
        if (ref) ref[((long)b * N_ + t) * H_ + hid] = (_Float16)hn;
    }
}

// ---------------- key GEMM: out(Mx128) = A(Mx128) * W^T + bias (f16 out) ----------------
// grid: M/16 * 2 waves; each wave 16(M) x 64(N), 4 k-steps, double-buffered
__global__ void gemm_k128_n128_bias_f16(const _Float16* __restrict__ A,
                                        const _Float16* __restrict__ W,   // 128x128 row-major
                                        const float* __restrict__ bias,
                                        _Float16* __restrict__ out, int M) {
    int lane  = threadIdx.x & 31;
    int wave  = blockIdx.x * (blockDim.x >> 5) + (threadIdx.x >> 5);
    int mBase = (wave >> 1) << 4;
    if (mBase >= M) return;                 // wave-uniform: EXEC stays all-ones
    int nBase = (wave & 1) << 6;
    int ncol  = nBase + (lane & 15);
    const _Float16* Arow = A + (size_t)(mBase + (lane & 15)) * H_;
    const _Float16* Wr0  = W + (size_t)(ncol     ) * H_;
    const _Float16* Wr1  = W + (size_t)(ncol + 16) * H_;
    const _Float16* Wr2  = W + (size_t)(ncol + 32) * H_;
    const _Float16* Wr3  = W + (size_t)(ncol + 48) * H_;

    v8f acc0 = {}, acc1 = {}, acc2 = {}, acc3 = {};
    v16h a  = load_frag(Arow, lane);
    v16h b0 = load_frag(Wr0, lane);
    v16h b1 = load_frag(Wr1, lane);
    v16h b2 = load_frag(Wr2, lane);
    v16h b3 = load_frag(Wr3, lane);
    #pragma unroll
    for (int kk = 0; kk < 4; ++kk) {
        v16h an, c0, c1, c2, c3;
        if (kk < 3) {
            const int k1 = (kk + 1) << 5;
            an = load_frag(Arow + k1, lane);
            c0 = load_frag(Wr0 + k1, lane);
            c1 = load_frag(Wr1 + k1, lane);
            c2 = load_frag(Wr2 + k1, lane);
            c3 = load_frag(Wr3 + k1, lane);
        }
        acc0 = WMMA_F16(a, b0, acc0);
        acc1 = WMMA_F16(a, b1, acc1);
        acc2 = WMMA_F16(a, b2, acc2);
        acc3 = WMMA_F16(a, b3, acc3);
        if (kk < 3) { a = an; b0 = c0; b1 = c1; b2 = c2; b3 = c3; }
    }
    int crow0 = mBase + ((lane >> 4) << 3);
    int ccol  = lane & 15;
    #pragma unroll
    for (int r = 0; r < 8; ++r) {
        _Float16* o = out + (size_t)(crow0 + r) * H_ + nBase + ccol;
        o[0]  = (_Float16)(acc0[r] + bias[nBase + ccol]);
        o[16] = (_Float16)(acc1[r] + bias[nBase + ccol + 16]);
        o[32] = (_Float16)(acc2[r] + bias[nBase + ccol + 32]);
        o[48] = (_Float16)(acc3[r] + bias[nBase + ccol + 48]);
    }
}

// ---------------- glimpse: softmax attention readout (one block per batch row) ----------------
__global__ void glimpse_kernel(const _Float16* __restrict__ h, const float* __restrict__ Wq,
                               const float* __restrict__ bq, const _Float16* __restrict__ u2,
                               const float* __restrict__ Vec, const float* __restrict__ mask,
                               float* __restrict__ query) {
    int b = blockIdx.x, tid = threadIdx.x;                  // 256 threads
    __shared__ float u1s[H_];
    __shared__ float us[N_];
    __shared__ float redv[256];
    __shared__ float s_max, s_sum;

    if (tid < H_) {
        const _Float16* hr = h + (size_t)b * H_;
        const float* wr = Wq + (size_t)tid * H_;
        float s = bq[tid];
        #pragma unroll 8
        for (int k = 0; k < H_; ++k) s += (float)hr[k] * wr[k];
        u1s[tid] = s;
    }
    __syncthreads();
    for (int n = tid; n < N_; n += 256) {
        const _Float16* ur = u2 + ((size_t)b * N_ + n) * H_;
        float s = 0.f;
        for (int g = 0; g < H_; ++g) s += tanhf(u1s[g] + (float)ur[g]) * Vec[g];
        us[n] = s - INF_ * mask[(size_t)b * N_ + n];
    }
    __syncthreads();
    redv[tid] = fmaxf(us[tid], us[tid + 256]);
    __syncthreads();
    for (int st = 128; st > 0; st >>= 1) {
        if (tid < st) redv[tid] = fmaxf(redv[tid], redv[tid + st]);
        __syncthreads();
    }
    if (tid == 0) s_max = redv[0];
    __syncthreads();
    float e0 = expf(us[tid] - s_max), e1 = expf(us[tid + 256] - s_max);
    us[tid] = e0; us[tid + 256] = e1; redv[tid] = e0 + e1;
    __syncthreads();
    for (int st = 128; st > 0; st >>= 1) {
        if (tid < st) redv[tid] += redv[tid + st];
        __syncthreads();
    }
    if (tid == 0) s_sum = redv[0];
    __syncthreads();
    if (tid < H_) {
        float inv = 1.f / s_sum, q = 0.f;
        const _Float16* base = u2 + (size_t)b * N_ * H_ + tid;
        for (int n = 0; n < N_; ++n) q += us[n] * (float)base[(size_t)n * H_];
        query[(size_t)b * H_ + tid] = q * inv;
    }
}

// ---------------- pointing: clipped logits, log-softmax, greedy argmax, state update ----------------
__global__ void point_kernel(const float* __restrict__ query, const float* __restrict__ Wq2,
                             const float* __restrict__ bq2, const _Float16* __restrict__ u2p,
                             const float* __restrict__ Vec2, float* __restrict__ mask,
                             const _Float16* __restrict__ emb, _Float16* __restrict__ dec_in,
                             int* __restrict__ pi, float* __restrict__ ll, int step) {
    int b = blockIdx.x, tid = threadIdx.x;                  // 256 threads
    __shared__ float u1s[H_];
    __shared__ float us[N_];
    __shared__ float redv[256];
    __shared__ int   redi[256];
    __shared__ int   s_nxt;
    __shared__ float s_max;

    if (tid < H_) {
        const float* qr = query + (size_t)b * H_;
        const float* wr = Wq2 + (size_t)tid * H_;
        float s = bq2[tid];
        #pragma unroll 8
        for (int k = 0; k < H_; ++k) s += qr[k] * wr[k];
        u1s[tid] = s;
    }
    __syncthreads();
    for (int n = tid; n < N_; n += 256) {
        const _Float16* ur = u2p + ((size_t)b * N_ + n) * H_;
        float s = 0.f;
        for (int g = 0; g < H_; ++g) s += tanhf(u1s[g] + (float)ur[g]) * Vec2[g];
        us[n] = CLIP_ * s - INF_ * mask[(size_t)b * N_ + n];
    }
    __syncthreads();
    // argmax (first max index) + max
    float v1 = us[tid]; int i1 = tid;
    float v2 = us[tid + 256];
    if (v2 > v1) { v1 = v2; i1 = tid + 256; }
    redv[tid] = v1; redi[tid] = i1;
    __syncthreads();
    for (int st = 128; st > 0; st >>= 1) {
        if (tid < st) {
            float a = redv[tid], cc = redv[tid + st];
            if (cc > a || (cc == a && redi[tid + st] < redi[tid])) {
                redv[tid] = cc; redi[tid] = redi[tid + st];
            }
        }
        __syncthreads();
    }
    if (tid == 0) { s_nxt = redi[0]; s_max = redv[0]; }
    __syncthreads();
    redv[tid] = expf(us[tid] - s_max) + expf(us[tid + 256] - s_max);
    __syncthreads();
    for (int st = 128; st > 0; st >>= 1) {
        if (tid < st) redv[tid] += redv[tid + st];
        __syncthreads();
    }
    if (tid == 0) {
        float lse = s_max + logf(redv[0]);
        int nxt = s_nxt;
        ll[b] += us[nxt] - lse;
        pi[b * 3 + step] = nxt;
        mask[(size_t)b * N_ + nxt] = 1.f;
    }
    __syncthreads();
    if (tid < H_) dec_in[(size_t)b * H_ + tid] = emb[((size_t)b * N_ + s_nxt) * H_ + tid];
}

__global__ void finalize_kernel(const int* __restrict__ pi, const float* __restrict__ ll,
                                float* __restrict__ out) {
    int t = blockIdx.x * blockDim.x + threadIdx.x;
    if (t < B_ * 3)           out[t] = (float)pi[t];
    else if (t < B_ * 3 + B_) out[t] = ll[t - B_ * 3];
}

// ---------------- host orchestration ----------------
extern "C" void kernel_launch(void* const* d_in, const int* in_sizes, int n_in,
                              void* d_out, int out_size, void* d_ws, size_t ws_size,
                              hipStream_t stream) {
    (void)in_sizes; (void)n_in; (void)out_size; (void)ws_size;
    const float* x        = (const float*)d_in[0];
    const float* red      = (const float*)d_in[1];
    const float* W_emb    = (const float*)d_in[2];
    const float* enc_Wih  = (const float*)d_in[3];
    const float* enc_Whh  = (const float*)d_in[4];
    const float* enc_bih  = (const float*)d_in[5];
    const float* enc_bhh  = (const float*)d_in[6];
    const float* dec_Wih  = (const float*)d_in[7];
    const float* dec_Whh  = (const float*)d_in[8];
    const float* dec_bih  = (const float*)d_in[9];
    const float* dec_bhh  = (const float*)d_in[10];
    const float* Wq    = (const float*)d_in[11];
    const float* bq    = (const float*)d_in[12];
    const float* Wref  = (const float*)d_in[13];
    const float* bref  = (const float*)d_in[14];
    const float* Wq2   = (const float*)d_in[15];
    const float* bq2   = (const float*)d_in[16];
    const float* Wref2 = (const float*)d_in[17];
    const float* bref2 = (const float*)d_in[18];
    const float* Vec   = (const float*)d_in[19];
    const float* Vec2  = (const float*)d_in[20];

    char* wsb = (char*)d_ws;
    size_t off = 0;
    auto alloc = [&](size_t bytes) -> char* {
        char* p = wsb + off;
        off += (bytes + 255) & ~(size_t)255;
        return p;
    };
    _Float16* emb    = (_Float16*)alloc((size_t)B_ * N_ * H_ * 2);   // 64 MB
    _Float16* ref    = (_Float16*)alloc((size_t)B_ * N_ * H_ * 2);   // 64 MB
    _Float16* u2     = (_Float16*)alloc((size_t)B_ * N_ * H_ * 2);   // 64 MB
    _Float16* u2p    = (_Float16*)alloc((size_t)B_ * N_ * H_ * 2);   // 64 MB
    _Float16* WcatE  = (_Float16*)alloc((size_t)G4_ * 256 * 2);
    _Float16* WcatD  = (_Float16*)alloc((size_t)G4_ * 256 * 2);
    float*    biasE  = (float*)   alloc((size_t)G4_ * 4);
    float*    biasD  = (float*)   alloc((size_t)G4_ * 4);
    _Float16* Wref16 = (_Float16*)alloc((size_t)H_ * H_ * 2);
    _Float16* Wre216 = (_Float16*)alloc((size_t)H_ * H_ * 2);
    _Float16* h      = (_Float16*)alloc((size_t)B_ * H_ * 2);
    float*    c      = (float*)   alloc((size_t)B_ * H_ * 4);
    _Float16* dec_in = (_Float16*)alloc((size_t)B_ * H_ * 2);
    float*    query  = (float*)   alloc((size_t)B_ * H_ * 4);
    float*    mask   = (float*)   alloc((size_t)B_ * N_ * 4);
    int*      pi     = (int*)     alloc((size_t)B_ * 3 * 4);
    float*    ll     = (float*)   alloc((size_t)B_ * 4);

    // ---- weight packing (f32 -> f16, concat [Wih|Whh], summed biases) ----
    pack_wcat<<<512, 256, 0, stream>>>(enc_Wih, enc_Whh, WcatE);
    pack_wcat<<<512, 256, 0, stream>>>(dec_Wih, dec_Whh, WcatD);
    add_bias_vec<<<2, 256, 0, stream>>>(enc_bih, enc_bhh, biasE, G4_);
    add_bias_vec<<<2, 256, 0, stream>>>(dec_bih, dec_bhh, biasD, G4_);
    f32_to_f16_kernel<<<64, 256, 0, stream>>>(Wref,  Wref16, H_ * H_);
    f32_to_f16_kernel<<<64, 256, 0, stream>>>(Wref2, Wre216, H_ * H_);

    // ---- embedding + initial decoder input + state init ----
    embed_kernel<<<(B_ * N_ * H_) / 256, 256, 0, stream>>>(x, W_emb, emb, B_ * N_);
    embed_kernel<<<(B_ * H_) / 256, 256, 0, stream>>>(red, W_emb, dec_in, B_);
    init_state<<<(B_ * N_) / 256, 256, 0, stream>>>(h, c, mask, ll, pi);

    // ---- encoder LSTM scan (512 sequential fused WMMA+cell steps) ----
    for (int t = 0; t < N_; ++t) {
        lstm_step_fused<<<32, 256, 0, stream>>>(emb + (size_t)t * H_, (long)N_ * H_,
                                                h, c, WcatE, biasE, ref, t);
    }

    // ---- attention keys (computed once) ----
    gemm_k128_n128_bias_f16<<<4096, 256, 0, stream>>>(ref, Wref16, bref,  u2,  B_ * N_);
    gemm_k128_n128_bias_f16<<<4096, 256, 0, stream>>>(ref, Wre216, bref2, u2p, B_ * N_);

    // ---- 3 greedy decode steps ----
    for (int s = 0; s < 3; ++s) {
        lstm_step_fused<<<32, 256, 0, stream>>>(dec_in, (long)H_, h, c, WcatD, biasD,
                                                (_Float16*)nullptr, 0);
        glimpse_kernel<<<B_, 256, 0, stream>>>(h, Wq, bq, u2, Vec, mask, query);
        point_kernel<<<B_, 256, 0, stream>>>(query, Wq2, bq2, u2p, Vec2, mask,
                                             emb, dec_in, pi, ll, s);
    }

    finalize_kernel<<<8, 256, 0, stream>>>(pi, ll, (float*)d_out);
}
